// MCNN_13950053777732
// MI455X (gfx1250) — compile-verified, matrix operand order checked by reading, not run on hardware
//
#include <hip/hip_runtime.h>

typedef __attribute__((ext_vector_type(16))) __bf16 v16bf;
typedef __attribute__((ext_vector_type(8)))  __bf16 v8bf;
typedef __attribute__((ext_vector_type(8)))  float  v8f;

#define B_   32
#define L_   2048
#define D_   256
#define T_   64
#define KW   3
#define ND   (KW * D_)      // 768
#define TILE_M 128
#define XS_ROWS (TILE_M + 2)

__device__ __forceinline__ unsigned short f2bf(float f) {
    union { __bf16 h; unsigned short u; } c;
    c.h = (__bf16)f;                     // native cvt path on gfx1250
    return c.u;
}

// ---------------------------------------------------------------------------
// Prep: kcT[col][t] = bf16(relu(C@W + b)) (transposed for contiguous B-frags),
//       Cb[t][d]    = bf16(C[t][d])
// ---------------------------------------------------------------------------
__global__ __launch_bounds__(256) void mcnn_prep_kernel(
        const float* __restrict__ C, const float* __restrict__ W,
        const float* __restrict__ bden,
        unsigned short* __restrict__ kcT, unsigned short* __restrict__ Cb) {
    const int t = blockIdx.x;        // 0..63
    const int tid = threadIdx.x;     // 0..255
    Cb[t * D_ + tid] = f2bf(C[t * D_ + tid]);
    const float* crow = C + t * D_;
    for (int rep = 0; rep < 3; ++rep) {
        const int col = tid + rep * 256;
        float acc = bden[col];
        for (int d = 0; d < D_; ++d) acc += crow[d] * W[d * ND + col];
        acc = acc > 0.f ? acc : 0.f;
        kcT[col * T_ + t] = f2bf(acc);
    }
}

// ---------------------------------------------------------------------------
// Fused: att = x@C^T (WMMA bf16), ki = att@kc (WMMA bf16), windowed combine.
// One workgroup = 128 consecutive positions of one batch; 8 waves, each owns
// a 16-row stripe. Dynamic LDS layout:
//   xs  : fp32 x tile with halo   (130*256*4 = 133120 B)
//   Cb  : bf16 C [64][256]        (32768 B)
//   Kb  : bf16 kcT [768][64]      (98304 B)
//   Ab  : bf16 att [128][64]      (16384 B)        total 280576 B < 320 KB
// ---------------------------------------------------------------------------
extern __shared__ char smem_raw[];

__global__ __launch_bounds__(256, 1) void mcnn_fused_kernel(
        const float* __restrict__ x,
        const unsigned short* __restrict__ Cb_g,
        const unsigned short* __restrict__ kcT_g,
        float* __restrict__ out) {
    float*          xs = (float*)smem_raw;
    unsigned short* Cb = (unsigned short*)(xs + XS_ROWS * D_);
    unsigned short* Kb = Cb + T_ * D_;
    unsigned short* Ab = Kb + ND * T_;

    const int b   = blockIdx.y;
    const int l0  = blockIdx.x * TILE_M;
    const int tid = threadIdx.x;

    // ---- cooperative staging ----
    for (int i = tid; i < XS_ROWS * D_; i += 256) {
        const int row = i >> 8, col = i & 255;
        const int g = l0 - 1 + row;
        xs[i] = (g >= 0 && g < L_) ? x[((size_t)b * L_ + g) * D_ + col] : 0.f;
    }
    for (int i = tid; i < T_ * D_; i += 256) Cb[i] = Cb_g[i];
    for (int i = tid; i < ND * T_; i += 256) Kb[i] = kcT_g[i];
    __syncthreads();

    const int lane = tid & 31;
    const int wave = tid >> 5;        // 0..7
    const int m0   = wave * 16;
    const int ln   = lane & 15;
    const int hi   = lane >> 4;

    // ================= GEMM1: att[16x64] = x[16x256] @ C^T =================
    v8f acc[4];
    #pragma unroll
    for (int nt = 0; nt < 4; ++nt) acc[nt] = (v8f){};

    for (int kt = 0; kt < 8; ++kt) {
        const int k0 = kt * 32;
        // A fragment: lane = row m0+ln, k-chunks {k0+hi*8.., k0+16+hi*8..}
        const float* rp = xs + (m0 + ln + 1) * D_ + k0 + hi * 8;   // 32B aligned
        v8f f0 = *reinterpret_cast<const v8f*>(rp);
        v8f f1 = *reinterpret_cast<const v8f*>(rp + 16);
        v8bf c0 = __builtin_convertvector(f0, v8bf);               // packed cvt
        v8bf c1 = __builtin_convertvector(f1, v8bf);
        v16bf a = __builtin_shufflevector(c0, c1, 0, 1, 2, 3, 4, 5, 6, 7,
                                                  8, 9, 10, 11, 12, 13, 14, 15);

        #pragma unroll
        for (int nt = 0; nt < 4; ++nt) {
            union { v16bf v; uint4 q[2]; } bb;
            const uint4* bp =
                reinterpret_cast<const uint4*>(Cb + (nt * 16 + ln) * D_ + k0 + hi * 8);
            bb.q[0] = bp[0];
            bb.q[1] = bp[2];   // +16 bf16 elements = +32 bytes
            acc[nt] = __builtin_amdgcn_wmma_f32_16x16x32_bf16(
                false, a, false, bb.v, (short)0, acc[nt], false, false);
        }
    }

    // scatter att (D-layout: VGPR v -> row v+8*hi, col = ln) as bf16
    #pragma unroll
    for (int nt = 0; nt < 4; ++nt) {
        union { v8bf v; unsigned short u[8]; } ac;
        ac.v = __builtin_convertvector(acc[nt], v8bf);
        #pragma unroll
        for (int v = 0; v < 8; ++v)
            Ab[(m0 + v + 8 * hi) * T_ + nt * 16 + ln] = ac.u[v];
    }
    __syncthreads();

    // ============ GEMM2 + window combine, ki never hits memory =============
    union { v16bf v; uint4 q[2]; } a2[2];
    #pragma unroll
    for (int kt = 0; kt < 2; ++kt) {
        const uint4* ap =
            reinterpret_cast<const uint4*>(Ab + (m0 + ln) * T_ + kt * 32 + hi * 8);
        a2[kt].q[0] = ap[0];
        a2[kt].q[1] = ap[2];
    }

    v8f outacc[16];
    #pragma unroll
    for (int dt = 0; dt < 16; ++dt) outacc[dt] = (v8f){};

    for (int dt = 0; dt < 16; ++dt) {
        #pragma unroll
        for (int j = 0; j < KW; ++j) {
            const int nbase = j * D_ + dt * 16;
            v8f kv = {};
            #pragma unroll
            for (int kt = 0; kt < 2; ++kt) {
                union { v16bf v; uint4 q[2]; } bb;
                const uint4* bp =
                    reinterpret_cast<const uint4*>(Kb + (nbase + ln) * T_ + kt * 32 + hi * 8);
                bb.q[0] = bp[0];
                bb.q[1] = bp[2];
                kv = __builtin_amdgcn_wmma_f32_16x16x32_bf16(
                    false, a2[kt].v, false, bb.v, (short)0, kv, false, false);
            }
            const int d = dt * 16 + ln;
            #pragma unroll
            for (int v = 0; v < 8; ++v) {
                const int row = m0 + v + 8 * hi;          // tile-local 0..127
                outacc[dt][v] += kv[v] * xs[(row + j) * D_ + d];
            }
        }
    }

    // ---- store out tile (fp32) ----
    #pragma unroll
    for (int dt = 0; dt < 16; ++dt) {
        const int d = dt * 16 + ln;
        #pragma unroll
        for (int v = 0; v < 8; ++v) {
            const int row = m0 + v + 8 * hi;
            out[((size_t)b * L_ + l0 + row) * D_ + d] = outacc[dt][v];
        }
    }
}

// ---------------------------------------------------------------------------
extern "C" void kernel_launch(void* const* d_in, const int* in_sizes, int n_in,
                              void* d_out, int out_size, void* d_ws, size_t ws_size,
                              hipStream_t stream) {
    (void)in_sizes; (void)n_in; (void)out_size; (void)ws_size;
    const float* x    = (const float*)d_in[0];
    const float* C    = (const float*)d_in[1];
    const float* W    = (const float*)d_in[2];
    const float* bden = (const float*)d_in[3];

    unsigned short* kcT = (unsigned short*)d_ws;     // 768*64 bf16 = 96 KB
    unsigned short* Cb  = kcT + ND * T_;             // 64*256 bf16 = 32 KB

    mcnn_prep_kernel<<<dim3(T_), dim3(256), 0, stream>>>(C, W, bden, kcT, Cb);

    const size_t smem = (size_t)XS_ROWS * D_ * sizeof(float)
                      + (size_t)T_ * D_ * 2
                      + (size_t)ND * T_ * 2
                      + (size_t)TILE_M * T_ * 2;     // 280576 bytes
    mcnn_fused_kernel<<<dim3(L_ / TILE_M, B_), dim3(256), smem, stream>>>(
        x, Cb, kcT, (float*)d_out);
}
